// HierarchicalSharedWindowMTL_49426483642690
// MI455X (gfx1250) — compile-verified, matrix operand order checked by reading, not run on hardware
//
#include <hip/hip_runtime.h>
#include <hip/hip_bf16.h>

// ---------------------------------------------------------------------------
// HierarchicalSharedWindowMTL: routed-expert MLP (256->256 relu -> 4) with
// ordinal-probs head. Grouped-GEMM over expert buckets using CDNA5 WMMA
// (v_wmma_f32_16x16x32_f16, f32 accumulate), wave32.
// ---------------------------------------------------------------------------

#define BTOT   32768
#define IN_DIM 256
#define HID    256
#define NEXP   8
#define KM1    4          // K-1 ordinal logits
#define TILES  (BTOT / 16)

typedef __attribute__((ext_vector_type(16))) _Float16 v16h;
typedef __attribute__((ext_vector_type(8)))  _Float16 v8h;
typedef __attribute__((ext_vector_type(8)))  float    v8f;

// ---------------- kernel 0: zero expert counters ----------------
__global__ void hswmtl_zero_counts(int* __restrict__ cnt) {
    if (threadIdx.x < NEXP) cnt[threadIdx.x] = 0;
}

// ---------------- kernel 1: bucket sample ids per expert ----------------
__global__ void hswmtl_bucket(const int* __restrict__ size_idx,
                              int* __restrict__ cnt,
                              int* __restrict__ list) {
    int i = blockIdx.x * blockDim.x + threadIdx.x;
    if (i < BTOT) {
        int e = size_idx[i];
        int pos = atomicAdd(&cnt[e], 1);
        list[e * BTOT + pos] = i;
    }
}

// ---------------- kernel 2: W1 [E][in][hid] f32 -> W1t [E][hid][in] f16 -----
__global__ void hswmtl_w1t(const float* __restrict__ W1,
                           _Float16* __restrict__ w1t) {
    int o = blockIdx.x * blockDim.x + threadIdx.x;   // 0 .. 8*256*256-1
    int e = o >> 16;
    int rem = o & 65535;
    int n = rem >> 8;      // output column (hid)
    int k = rem & 255;     // input dim
    w1t[o] = (_Float16)W1[(e << 16) + (k << 8) + n];
}

// ---------------- kernel 3: grouped GEMM + head ----------------
__global__ __launch_bounds__(256)
void hswmtl_moe(const float* __restrict__ x,
                const float* __restrict__ b1,
                const float* __restrict__ W2,
                const float* __restrict__ b2,
                const int* __restrict__ cnt,
                const int* __restrict__ list,
                const _Float16* __restrict__ w1t,
                float* __restrict__ out) {
    const int e    = blockIdx.x / TILES;
    const int tile = blockIdx.x % TILES;
    const int c    = cnt[e];
    const int row0 = tile * 16;
    if (row0 >= c) return;                 // uniform per block
    const int valid = (c - row0 < 16) ? (c - row0) : 16;

    __shared__ _Float16 xs[16][264];       // f16 tile of x rows (+pad)
    __shared__ float    hs[16][HID];       // relu hidden activations
    __shared__ float    lg[16][KM1];       // logits
    __shared__ int      sid[16];           // gathered sample ids (-1 invalid)

    const int tid = threadIdx.x;

    if (tid < 16)
        sid[tid] = (tid < valid) ? list[e * BTOT + row0 + tid] : -1;
    __syncthreads();

    // ---- stage x rows -> f16 in LDS (16 threads per row, 16 elems each) ----
    {
        int r = tid >> 4;
        int cb = (tid & 15) * 16;
        int s = sid[r];
        if (s >= 0) {
            const float* xr = x + (size_t)s * IN_DIM + cb;
            #pragma unroll
            for (int j = 0; j < 16; ++j) xs[r][cb + j] = (_Float16)xr[j];
        } else {
            #pragma unroll
            for (int j = 0; j < 16; ++j) xs[r][cb + j] = (_Float16)0.0f;
        }
    }
    __syncthreads();

    // ---- layer 1: 16 x 256 tile via WMMA f32 <- f16*f16, K = 256 ----
    const int lane = tid & 31;
    const int wave = tid >> 5;             // 8 waves -> 8 * 32 cols = 256
    const int lm   = lane & 15;
    const int o    = (lane < 16) ? 0 : 8;  // K-half select per lane group
    const int n0   = wave * 32;

    v8f acc0 = {}; v8f acc1 = {};
    const _Float16* bt0 = w1t + ((size_t)e << 16) + (size_t)(n0 + lm) * IN_DIM;
    const _Float16* bt1 = w1t + ((size_t)e << 16) + (size_t)(n0 + 16 + lm) * IN_DIM;
    const _Float16* ar  = &xs[lm][0];

    #pragma unroll
    for (int kc = 0; kc < 8; ++kc) {
        const int k0 = kc * 32;
        v8h alo  = *(const v8h*)(ar  + k0 + o);
        v8h ahi  = *(const v8h*)(ar  + k0 + 16 + o);
        v8h b0lo = *(const v8h*)(bt0 + k0 + o);
        v8h b0hi = *(const v8h*)(bt0 + k0 + 16 + o);
        v8h b1lo = *(const v8h*)(bt1 + k0 + o);
        v8h b1hi = *(const v8h*)(bt1 + k0 + 16 + o);
        v16h a, fb0, fb1;
        #pragma unroll
        for (int i = 0; i < 8; ++i) {
            a[i] = alo[i];  a[i + 8] = ahi[i];
            fb0[i] = b0lo[i]; fb0[i + 8] = b0hi[i];
            fb1[i] = b1lo[i]; fb1[i + 8] = b1hi[i];
        }
        acc0 = __builtin_amdgcn_wmma_f32_16x16x32_f16(
                   false, a, false, fb0, (short)0, acc0, false, false);
        acc1 = __builtin_amdgcn_wmma_f32_16x16x32_f16(
                   false, a, false, fb1, (short)0, acc1, false, false);
    }

    // ---- bias + relu, write hidden tile to LDS ----
    {
        const int moff = (lane < 16) ? 0 : 8;   // C/D layout: vgpr r -> M=r(+8)
        const float bias0 = b1[e * HID + n0 + lm];
        const float bias1 = b1[e * HID + n0 + 16 + lm];
        #pragma unroll
        for (int r = 0; r < 8; ++r) {
            const int m = r + moff;
            float h0 = acc0[r] + bias0;
            float h1 = acc1[r] + bias1;
            hs[m][n0 + lm]      = h0 > 0.0f ? h0 : 0.0f;
            hs[m][n0 + 16 + lm] = h1 > 0.0f ? h1 : 0.0f;
        }
    }
    __syncthreads();

    // ---- layer 2: h(16x256) @ W2[e](256x4) + b2 ----
    if (tid < 16 * KM1) {
        const int m  = tid >> 2;
        const int kk = tid & 3;
        float s = b2[e * KM1 + kk];
        const float* w2p = W2 + (size_t)e * HID * KM1 + kk;
        #pragma unroll 8
        for (int i = 0; i < HID; ++i) s += hs[m][i] * w2p[i * KM1];
        lg[m][kk] = s;
        const int sm = sid[m];
        if (sm >= 0) out[(size_t)sm * KM1 + kk] = s;   // logits output
    }
    __syncthreads();

    // ---- ordinal logits -> probs ----
    if (tid < 16) {
        const int sm = sid[tid];
        if (sm >= 0) {
            float q0 = 1.0f / (1.0f + __expf(-lg[tid][0]));
            float q1 = 1.0f / (1.0f + __expf(-lg[tid][1]));
            float q2 = 1.0f / (1.0f + __expf(-lg[tid][2]));
            float q3 = 1.0f / (1.0f + __expf(-lg[tid][3]));
            float p[5] = {1.0f - q0, q0 - q1, q1 - q2, q2 - q3, q3};
            float ssum = 0.0f;
            #pragma unroll
            for (int j = 0; j < 5; ++j) { p[j] = fmaxf(p[j], 1e-8f); ssum += p[j]; }
            const float inv = 1.0f / fmaxf(ssum, 1e-8f);
            float* pd = out + (size_t)BTOT * KM1 + (size_t)sm * 5;
            #pragma unroll
            for (int j = 0; j < 5; ++j) pd[j] = p[j] * inv;
        }
    }
}

// ---------------------------------------------------------------------------
extern "C" void kernel_launch(void* const* d_in, const int* in_sizes, int n_in,
                              void* d_out, int out_size, void* d_ws, size_t ws_size,
                              hipStream_t stream) {
    const float* x        = (const float*)d_in[0];
    const int*   size_idx = (const int*)  d_in[1];
    const float* W1       = (const float*)d_in[2];
    const float* b1       = (const float*)d_in[3];
    const float* W2       = (const float*)d_in[4];
    const float* b2       = (const float*)d_in[5];
    float*       out      = (float*)d_out;

    // workspace layout
    char* ws = (char*)d_ws;
    int*      cnt  = (int*)(ws);                                  //    32 B
    int*      list = (int*)(ws + 1024);                           //   1 MiB
    _Float16* w1t  = (_Float16*)(ws + 1024 + (size_t)NEXP * BTOT * 4); // 1 MiB

    hswmtl_zero_counts<<<1, 32, 0, stream>>>(cnt);
    hswmtl_bucket<<<(BTOT + 255) / 256, 256, 0, stream>>>(size_idx, cnt, list);
    hswmtl_w1t<<<(NEXP * IN_DIM * HID) / 256, 256, 0, stream>>>(W1, w1t);
    hswmtl_moe<<<NEXP * TILES, 256, 0, stream>>>(x, b1, W2, b2, cnt, list, w1t, out);
}
